// EdgeAwareGraphTransformerLayer_52836687676055
// MI455X (gfx1250) — matrix operand order repaired and checked by reading, not compiled, not run on hardware
//
#include <hip/hip_runtime.h>
#include <hip/hip_bf16.h>

// ---------------- problem constants ----------------
#define GG   32
#define NPG  512
#define NTOT (GG * NPG)      // 16384
#define DD   256
#define HH   8
#define DHH  32
#define EPG  8192
#define ETOT (GG * EPG)      // 262144
#define FFD  1024

// ---------------- CDNA5 WMMA / TDM types ----------------
typedef __attribute__((ext_vector_type(16))) __bf16        v16bf;
typedef __attribute__((ext_vector_type(8)))  __bf16        v8bf;
typedef __attribute__((ext_vector_type(8)))  float         v8f;
typedef __attribute__((ext_vector_type(4)))  unsigned int  u32x4;
typedef __attribute__((ext_vector_type(8)))  int           i32x8;
typedef __attribute__((ext_vector_type(4)))  int           i32x4;

__device__ __forceinline__ v16bf cat8(v8bf lo, v8bf hi) {
    return __builtin_shufflevector(lo, hi, 0,1,2,3,4,5,6,7,8,9,10,11,12,13,14,15);
}

// A-fragment (16x32 bf16, M x K): lane row = lane&15; lanes<16 take K
// [0..15]; lanes>=16 take [16..31]. base points at row start.
__device__ __forceinline__ v16bf load_afrag(const __bf16* base, int lane) {
    int lo = (lane & 16) ? 16 : 0;
    v8bf c0 = *(const v8bf*)(base + lo);
    v8bf c1 = *(const v8bf*)(base + lo + 8);
    return cat8(c0, c1);
}

// B-fragment (32x16 bf16, K x N): lane col = lane&15; lanes<16 hold K 0..15,
// lanes>=16 hold K 16..31. p points at the 16 contiguous bf16 for this lane.
__device__ __forceinline__ v16bf load_bfrag16(const __bf16* p) {
    v8bf c0 = *(const v8bf*)(p);
    v8bf c1 = *(const v8bf*)(p + 8);
    return cat8(c0, c1);
}

__device__ __forceinline__ v8f wmma_bf16(v16bf a, v16bf b, v8f c) {
    return __builtin_amdgcn_wmma_f32_16x16x32_bf16(
        false, a, false, b, (short)0, c, false, false);
}

// ---------------- Tensor Data Mover: 2-D bf16 tile -> LDS ----------------
// D# per CDNA5 ISA: group0 {count=1, lds_addr,
// global_addr, type=2}; group1 {data_size=1(2B), tensor dims, tile dims,
// dim0 stride}; groups 2/3 zero (<=2-D tile). Tracked by TENSORcnt.
__device__ __forceinline__ void tdm_load_2d(void* lds_ptr, const __bf16* gptr,
                                            unsigned tile_d0, unsigned tile_d1,
                                            unsigned long long stride_elems) {
    unsigned long long ga = (unsigned long long)(size_t)gptr;
    unsigned lds_addr = (unsigned)(size_t)lds_ptr;
    u32x4 g0;
    g0[0] = 1u;                                              // count=1, user D#
    g0[1] = lds_addr;
    g0[2] = (unsigned)(ga & 0xffffffffu);
    g0[3] = (unsigned)((ga >> 32) & 0x1ffffffu) | (2u << 30); // type=2 (image)
    i32x8 g1;
    g1[0] = (int)(1u << 16);                                  // data_size=2B
    g1[1] = (int)((tile_d0 & 0xffffu) << 16);                 // tensor_dim0 = tile_d0
    g1[2] = (int)((tile_d0 >> 16) | ((tile_d1 & 0xffffu) << 16)); // tensor_dim1
    g1[3] = (int)((tile_d1 >> 16) | ((tile_d0 & 0xffffu) << 16)); // tile_dim0
    g1[4] = (int)(tile_d1 & 0xffffu);                         // tile_dim1, tile_dim2=0
    g1[5] = (int)(stride_elems & 0xffffffffu);                // dim0 stride lo
    g1[6] = (int)((stride_elems >> 32) & 0xffffu);            // dim0 stride hi
    g1[7] = 0;
    i32x4 gz = {0, 0, 0, 0};
#if __clang_major__ >= 23
    i32x8 gz8 = {0, 0, 0, 0, 0, 0, 0, 0};
    __builtin_amdgcn_tensor_load_to_lds(g0, g1, gz, gz, gz8, 0);
#else
    __builtin_amdgcn_tensor_load_to_lds(g0, g1, gz, gz, 0);
#endif
}

// ---------------- small utility kernels ----------------
__global__ void zero_kernel(float4* p, long n4) {
    long i = (long)blockIdx.x * blockDim.x + threadIdx.x;
    long stride = (long)gridDim.x * blockDim.x;
    float4 z; z.x = 0.f; z.y = 0.f; z.z = 0.f; z.w = 0.f;
    for (; i < n4; i += stride) p[i] = z;
}

// fp32 [R][C] -> bf16 [C][R]   (weight transpose + downconvert)
__global__ void wtrans_kernel(const float* __restrict__ W, __bf16* __restrict__ Wt,
                              int R, int C) {
    int tid = blockIdx.x * blockDim.x + threadIdx.x;
    if (tid >= R * C) return;
    int r = tid / C, c = tid % C;
    Wt[(size_t)c * R + r] = (__bf16)W[tid];
}

// v bf16 [N][256] -> vt bf16 [G][H*DH][NPG]
__global__ void vtrans_kernel(const __bf16* __restrict__ v, __bf16* __restrict__ vt) {
    int tid = blockIdx.x * blockDim.x + threadIdx.x;
    if (tid >= NTOT * DD) return;
    int n = tid >> 8, f = tid & 255;
    int g = n >> 9, l = n & 511;
    vt[((size_t)g * 256 + f) * NPG + l] = v[tid];
}

// ---------------- layernorm kernels ----------------
__device__ __forceinline__ float block_sum256(float v, float* red) {
    int lane = threadIdx.x & 31, w = threadIdx.x >> 5;
    #pragma unroll
    for (int o = 16; o > 0; o >>= 1) v += __shfl_down(v, o, 32);
    if (lane == 0) red[w] = v;
    __syncthreads();
    float tot = 0.f;
    #pragma unroll
    for (int i = 0; i < 8; i++) tot += red[i];
    __syncthreads();
    return tot;
}

__global__ __launch_bounds__(256) void ln1_kernel(const float* __restrict__ x,
                                                  const float* __restrict__ g,
                                                  const float* __restrict__ b,
                                                  __bf16* __restrict__ xn) {
    __shared__ float red[8];
    int row = blockIdx.x, t = threadIdx.x;
    size_t idx = (size_t)row * DD + t;
    float v = x[idx];
    float mean = block_sum256(v, red) * (1.f / 256.f);
    float d = v - mean;
    float var = block_sum256(d * d, red) * (1.f / 256.f);
    float r = rsqrtf(var + 1e-5f);
    xn[idx] = (__bf16)(d * r * g[t] + b[t]);
}

// x1 = x + attn ; h2 = LN2(x1)
__global__ __launch_bounds__(256) void ln2res_kernel(const float* __restrict__ x,
                                                     const float* __restrict__ attn,
                                                     const float* __restrict__ g,
                                                     const float* __restrict__ b,
                                                     float* __restrict__ x1,
                                                     __bf16* __restrict__ h2) {
    __shared__ float red[8];
    int row = blockIdx.x, t = threadIdx.x;
    size_t idx = (size_t)row * DD + t;
    float v = x[idx] + attn[idx];
    x1[idx] = v;
    float mean = block_sum256(v, red) * (1.f / 256.f);
    float d = v - mean;
    float var = block_sum256(d * d, red) * (1.f / 256.f);
    float r = rsqrtf(var + 1e-5f);
    h2[idx] = (__bf16)(d * r * g[t] + b[t]);
}

// ---------------- edge bias kernels ----------------
__global__ void edge_bias_kernel(const int* __restrict__ ei, const float* __restrict__ ea,
                                 const float* __restrict__ wep, const float* __restrict__ bep,
                                 const float* __restrict__ weg, const float* __restrict__ beg,
                                 float* __restrict__ bias) {
    int e = blockIdx.x * blockDim.x + threadIdx.x;
    if (e >= ETOT) return;
    int s = ei[e], d = ei[ETOT + e];
    int g = s >> 9, ls = s & 511, ld = d & 511;
    float a0 = ea[2 * e], a1 = ea[2 * e + 1];
    #pragma unroll
    for (int h = 0; h < HH; h++) {
        float p  = a0 * wep[h] + a1 * wep[HH + h] + bep[h];
        float qg = a0 * weg[h] + a1 * weg[HH + h] + beg[h];
        float sig = 1.f / (1.f + __expf(-qg));
        atomicAdd(&bias[(((size_t)g * HH + h) * NPG + ls) * NPG + ld], p * sig);
    }
}

// faithful replication of the graph-0-only rel_pos adjacency
__global__ void aflag_kernel(const int* __restrict__ ei, unsigned char* __restrict__ af) {
    int e = blockIdx.x * blockDim.x + threadIdx.x;
    if (e >= ETOT) return;
    int s = ei[e], d = ei[ETOT + e];
    if (s < NPG && d < NPG) af[s * NPG + d] = 1;
}

__global__ void relpos_kernel(const unsigned char* __restrict__ af,
                              const float* __restrict__ rel,
                              float* __restrict__ bias) {
    int tid = blockIdx.x * blockDim.x + threadIdx.x; // 512*512
    if (tid >= NPG * NPG) return;
    if (!af[tid]) return;
    int qi = tid >> 9, ki = tid & 511;
    #pragma unroll
    for (int h = 0; h < HH; h++) {
        size_t bi = (((size_t)h) * NPG + qi) * NPG + ki; // graph 0 only
        bias[bi] = bias[bi] + rel[h];
    }
}

// ---------------- bf16 WMMA GEMM with TDM-staged weight panel ----------------
// out = act(A[MxK] @ Bt[NxK]^T + bias[N]) (+ resid)  ; per-wave 16x64 tile.
// The 64-row Bt panel is shared by all 8 waves -> TDM double-buffered 64x64
// chunks into LDS, DMA of chunk c+1 overlapped with WMMA on chunk c.
template <bool WB16, bool RELU, bool RES>
__global__ __launch_bounds__(256) void gemm_bf16_kernel(
    const __bf16* __restrict__ A, const __bf16* __restrict__ Bt,
    const float* __restrict__ bias, const float* __restrict__ resid,
    float* __restrict__ outF, __bf16* __restrict__ outB,
    int M, int Ncols, int K) {
    __shared__ __bf16 bT[2][64][64];  // 16 KB double buffer
    int lane = threadIdx.x & 31, wave = threadIdx.x >> 5;
    int m0 = blockIdx.y * 128 + wave * 16;
    int n0 = blockIdx.x * 64;
    int arow = m0 + (lane & 15);
    const __bf16* Abase = A + (size_t)arow * K;
    int klo = (lane & 16) ? 16 : 0;

    v8f acc[4];
    v8f zf = {0.f, 0.f, 0.f, 0.f, 0.f, 0.f, 0.f, 0.f};
    #pragma unroll
    for (int t = 0; t < 4; t++) acc[t] = zf;

    const __bf16* Bpanel = Bt + (size_t)n0 * K;
    int NC = K >> 6;                 // 64-wide k chunks
    if (wave == 0)
        tdm_load_2d(&bT[0][0][0], Bpanel, 64, 64, (unsigned long long)K);

    for (int c = 0; c < NC; ++c) {
        int buf = c & 1;
        if (wave == 0) {
            if (c + 1 < NC) {
                tdm_load_2d(&bT[buf ^ 1][0][0], Bpanel + (size_t)(c + 1) * 64,
                            64, 64, (unsigned long long)K);
                __builtin_amdgcn_s_wait_tensorcnt((short)1);
            } else {
                __builtin_amdgcn_s_wait_tensorcnt((short)0);
            }
        }
        __syncthreads();
        #pragma unroll
        for (int s = 0; s < 2; ++s) {
            int kk = c * 64 + s * 32;
            if (kk + 32 < K)
                __builtin_prefetch(Abase + kk + 32, 0, 1); // global_prefetch_b8
            v16bf a = load_afrag(Abase + kk, lane);
            #pragma unroll
            for (int t = 0; t < 4; t++) {
                const __bf16* bp = &bT[buf][t * 16 + (lane & 15)][s * 32 + klo];
                v16bf b = load_bfrag16(bp);
                acc[t] = wmma_bf16(a, b, acc[t]);
            }
        }
        __syncthreads();
    }

    int rbase = m0 + ((lane & 16) >> 1); // lanes>=16 hold rows +8
    #pragma unroll
    for (int t = 0; t < 4; t++) {
        int col = n0 + t * 16 + (lane & 15);
        float bv = bias[col];
        #pragma unroll
        for (int i = 0; i < 8; i++) {
            float val = acc[t][i] + bv;
            if (RELU) val = fmaxf(val, 0.f);
            size_t idx = (size_t)(rbase + i) * Ncols + col;
            if (RES) val += resid[idx];
            if (WB16) outB[idx] = (__bf16)val;
            else      outF[idx] = val;
        }
    }
}

// ---------------- fused flash attention (per g,h,64-row q tile) ----------------
// All 4 waves share the same (g,h,kt) K/V tiles -> TDM double-buffered 32x32
// tiles into LDS; fragments come from ds loads; P transposed through LDS.
__global__ __launch_bounds__(128) void attn_kernel(
    const __bf16* __restrict__ q, const __bf16* __restrict__ k,
    const __bf16* __restrict__ vt, const float* __restrict__ bias,
    float* __restrict__ attnout) {
    int lane = threadIdx.x & 31, wave = threadIdx.x >> 5;
    int bid = blockIdx.x;                 // g*64 + h*8 + qt
    int qt = bid & 7, h = (bid >> 3) & 7, g = bid >> 6;
    int q0 = qt * 64 + wave * 16;
    int col = lane & 15;
    int hsel = (lane & 16) ? 1 : 0;       // which 8-row half this lane owns

    __shared__ __bf16 kT[2][32][32];      // [buf][node][dh]   4 KB
    __shared__ __bf16 vT[2][32][32];      // [buf][dh][node]   4 KB
    __shared__ __bf16 pT[4][16][32];      // per-wave P tile   4 KB

    const float scale = 0.17677669529663687f; // 1/sqrt(32)

    // Q A-fragment (16 q-rows x 32 dh)
    int arow = q0 + (lane & 15);
    const __bf16* qbase = q + ((size_t)(g * NPG + arow)) * DD + h * DHH;
    v16bf qa = load_afrag(qbase, lane);

    v8f zf = {0.f, 0.f, 0.f, 0.f, 0.f, 0.f, 0.f, 0.f};
    v8f acc0 = zf, acc1 = zf;
    float m[8], l[8];
    #pragma unroll
    for (int i = 0; i < 8; i++) { m[i] = -1e30f; l[i] = 0.f; }

    size_t ghb = (size_t)(g * HH + h) * NPG;
    const __bf16* Kbase = k + ((size_t)(g * NPG)) * DD + h * DHH; // [node][dh], stride DD
    const __bf16* Vbase = vt + ((size_t)(g * HH + h) * DHH) * NPG; // [dh][node], stride NPG

    if (wave == 0) {
        tdm_load_2d(&kT[0][0][0], Kbase, 32, 32, (unsigned long long)DD);
        tdm_load_2d(&vT[0][0][0], Vbase, 32, 32, (unsigned long long)NPG);
    }

    for (int kt = 0, c = 0; kt < NPG; kt += 32, ++c) {
        int buf = c & 1;
        if (wave == 0) {
            if (kt + 32 < NPG) {
                tdm_load_2d(&kT[buf ^ 1][0][0], Kbase + (size_t)(kt + 32) * DD,
                            32, 32, (unsigned long long)DD);
                tdm_load_2d(&vT[buf ^ 1][0][0], Vbase + kt + 32,
                            32, 32, (unsigned long long)NPG);
                __builtin_amdgcn_s_wait_tensorcnt((short)2);
            } else {
                __builtin_amdgcn_s_wait_tensorcnt((short)0);
            }
        }
        __syncthreads();  // K/V tiles visible

        // scores: B[kd][n] = kT[buf][tile*16+col][kd]
        v8f c0 = wmma_bf16(qa, load_bfrag16(&kT[buf][col][hsel * 16]), zf);
        v8f c1 = wmma_bf16(qa, load_bfrag16(&kT[buf][16 + col][hsel * 16]), zf);

        float p0[8], p1[8], mn[8], al[8], rs[8];
        #pragma unroll
        for (int i = 0; i < 8; i++) {
            size_t bi = (ghb + q0 + i + hsel * 8) * NPG + kt + col;
            p0[i] = c0[i] * scale + bias[bi];
            p1[i] = c1[i] * scale + bias[bi + 16];
            mn[i] = fmaxf(p0[i], p1[i]);
        }
        #pragma unroll
        for (int o = 1; o < 16; o <<= 1) {
            #pragma unroll
            for (int i = 0; i < 8; i++) mn[i] = fmaxf(mn[i], __shfl_xor(mn[i], o, 32));
        }
        #pragma unroll
        for (int i = 0; i < 8; i++) {
            float mnew = fmaxf(m[i], mn[i]);
            al[i] = __expf(m[i] - mnew);
            m[i] = mnew;
            p0[i] = __expf(p0[i] - mnew);
            p1[i] = __expf(p1[i] - mnew);
            rs[i] = p0[i] + p1[i];
        }
        #pragma unroll
        for (int o = 1; o < 16; o <<= 1) {
            #pragma unroll
            for (int i = 0; i < 8; i++) rs[i] += __shfl_xor(rs[i], o, 32);
        }
        #pragma unroll
        for (int i = 0; i < 8; i++) {
            l[i] = l[i] * al[i] + rs[i];
            acc0[i] *= al[i];
            acc1[i] *= al[i];
        }

        // C-layout -> A-layout transpose of P through LDS
        __syncthreads();
        #pragma unroll
        for (int i = 0; i < 8; i++) {
            int rl = i + hsel * 8;
            pT[wave][rl][col]      = (__bf16)p0[i];
            pT[wave][rl][16 + col] = (__bf16)p1[i];
        }
        __syncthreads();
        v16bf pa = load_afrag(&pT[wave][lane & 15][0], lane);

        acc0 = wmma_bf16(pa, load_bfrag16(&vT[buf][col][hsel * 16]), acc0);
        acc1 = wmma_bf16(pa, load_bfrag16(&vT[buf][16 + col][hsel * 16]), acc1);

        __syncthreads();  // reads of buf done before next TDM overwrites
    }

    #pragma unroll
    for (int i = 0; i < 8; i++) {
        int row = q0 + i + hsel * 8;
        float inv = 1.f / l[i];
        size_t ob = ((size_t)(g * NPG + row)) * DD + h * DHH + col;
        attnout[ob]      = acc0[i] * inv;
        attnout[ob + 16] = acc1[i] * inv;
    }
}

// ---------------- launcher ----------------
extern "C" void kernel_launch(void* const* d_in, const int* in_sizes, int n_in,
                              void* d_out, int out_size, void* d_ws, size_t ws_size,
                              hipStream_t stream) {
    const float* x     = (const float*)d_in[0];
    const int*   ei    = (const int*)d_in[1];
    const float* ea    = (const float*)d_in[2];
    const float* ln1g  = (const float*)d_in[3];
    const float* ln1b  = (const float*)d_in[4];
    const float* wq    = (const float*)d_in[5];
    const float* bq    = (const float*)d_in[6];
    const float* wk    = (const float*)d_in[7];
    const float* bk    = (const float*)d_in[8];
    const float* wv    = (const float*)d_in[9];
    const float* bv    = (const float*)d_in[10];
    const float* wep   = (const float*)d_in[11];
    const float* bep   = (const float*)d_in[12];
    const float* weg   = (const float*)d_in[13];
    const float* beg   = (const float*)d_in[14];
    const float* rel   = (const float*)d_in[15];
    const float* w1    = (const float*)d_in[16];
    const float* b1    = (const float*)d_in[17];
    const float* w2    = (const float*)d_in[18];
    const float* b2    = (const float*)d_in[19];
    const float* ln2g  = (const float*)d_in[20];
    const float* ln2b  = (const float*)d_in[21];
    float* out = (float*)d_out;

    char* ws = (char*)d_ws;
    size_t off = 0;
    auto alloc = [&](size_t bytes) -> char* {
        off = (off + 255) & ~(size_t)255;
        char* p = ws + off;
        off += bytes;
        return p;
    };
    const size_t ND = (size_t)NTOT * DD;
    __bf16* xn   = (__bf16*)alloc(ND * 2);
    __bf16* qb   = (__bf16*)alloc(ND * 2);
    __bf16* kb   = (__bf16*)alloc(ND * 2);
    __bf16* vb_  = (__bf16*)alloc(ND * 2);
    __bf16* vtb  = (__bf16*)alloc(ND * 2);
    __bf16* wqt  = (__bf16*)alloc((size_t)DD * DD * 2);
    __bf16* wkt  = (__bf16*)alloc((size_t)DD * DD * 2);
    __bf16* wvt  = (__bf16*)alloc((size_t)DD * DD * 2);
    __bf16* w1t  = (__bf16*)alloc((size_t)DD * FFD * 2);
    __bf16* w2t  = (__bf16*)alloc((size_t)FFD * DD * 2);
    float*  bias = (float*)alloc((size_t)GG * HH * NPG * NPG * 4);
    unsigned char* af = (unsigned char*)alloc((size_t)NPG * NPG);
    float*  attn = (float*)alloc(ND * 4);
    float*  x1   = (float*)alloc(ND * 4);
    __bf16* h2   = (__bf16*)alloc(ND * 2);
    __bf16* ff1  = (__bf16*)alloc((size_t)NTOT * FFD * 2);

    // weight transpose + bf16 downconvert
    wtrans_kernel<<<(DD * DD + 255) / 256, 256, 0, stream>>>(wq, wqt, DD, DD);
    wtrans_kernel<<<(DD * DD + 255) / 256, 256, 0, stream>>>(wk, wkt, DD, DD);
    wtrans_kernel<<<(DD * DD + 255) / 256, 256, 0, stream>>>(wv, wvt, DD, DD);
    wtrans_kernel<<<(DD * FFD + 255) / 256, 256, 0, stream>>>(w1, w1t, DD, FFD);
    wtrans_kernel<<<(FFD * DD + 255) / 256, 256, 0, stream>>>(w2, w2t, FFD, DD);

    // LN1
    ln1_kernel<<<NTOT, 256, 0, stream>>>(x, ln1g, ln1b, xn);

    // zero bias tensor + adjacency flags
    zero_kernel<<<8192, 256, 0, stream>>>((float4*)bias, (long)GG * HH * NPG * NPG / 4);
    zero_kernel<<<64, 256, 0, stream>>>((float4*)af, (long)(NPG * NPG) / 16);

    // Q/K/V projections (bf16 out)
    dim3 gqkv(DD / 64, NTOT / 128);
    gemm_bf16_kernel<true, false, false><<<gqkv, 256, 0, stream>>>(xn, wqt, bq, nullptr, nullptr, qb, NTOT, DD, DD);
    gemm_bf16_kernel<true, false, false><<<gqkv, 256, 0, stream>>>(xn, wkt, bk, nullptr, nullptr, kb, NTOT, DD, DD);
    gemm_bf16_kernel<true, false, false><<<gqkv, 256, 0, stream>>>(xn, wvt, bv, nullptr, nullptr, vb_, NTOT, DD, DD);

    // V -> [g][h*dh][node]
    vtrans_kernel<<<(int)(ND / 256), 256, 0, stream>>>(vb_, vtb);

    // edge-conditioned bias (atomics) + graph-0 rel_pos adjacency
    edge_bias_kernel<<<ETOT / 256, 256, 0, stream>>>(ei, ea, wep, bep, weg, beg, bias);
    aflag_kernel<<<ETOT / 256, 256, 0, stream>>>(ei, af);
    relpos_kernel<<<(NPG * NPG) / 256, 256, 0, stream>>>(af, rel, bias);

    // fused attention
    attn_kernel<<<GG * HH * (NPG / 64), 128, 0, stream>>>(qb, kb, vtb, bias, attn);

    // residual + LN2
    ln2res_kernel<<<NTOT, 256, 0, stream>>>(x, attn, ln2g, ln2b, x1, h2);

    // FFN
    dim3 gff1(FFD / 64, NTOT / 128);
    gemm_bf16_kernel<true, true, false><<<gff1, 256, 0, stream>>>(h2, w1t, b1, nullptr, nullptr, ff1, NTOT, FFD, DD);
    dim3 gff2(DD / 64, NTOT / 128);
    gemm_bf16_kernel<false, false, true><<<gff2, 256, 0, stream>>>(ff1, w2t, b2, x1, out, nullptr, NTOT, DD, FFD);
}
